// GNN_Layer_79508434583745
// MI455X (gfx1250) — compile-verified, matrix operand order checked by reading, not run on hardware
//
#include <hip/hip_runtime.h>
#include <hip/hip_bf16.h>

typedef __attribute__((ext_vector_type(16))) _Float16 v16h;
typedef __attribute__((ext_vector_type(8)))  float    v8f;

#define NNODE 100000
#define NEDGE 1600000
// dims: in=64 edge=64 M=64 U=128 out=128
// edge layer1: K=128 (KC=4), OUT=64 (NT=4); edge layer2: K=64 (KC=2), OUT=64 (NT=4)
// node layer1: K=128 (KC=4), OUT=128 (NT=8); node layer2: K=128 (KC=4), OUT=128 (NT=8)

// ---------------------------------------------------------------------------
// Weight repack: A-fragment layout for V_WMMA_F32_16X16X32_F16.
// A (16x32 f16): lane L holds row M = L&15; half h -> K = (h>>3)*16 + (L>>4)*8 + (h&7).
// Packed index: [((n*KC + kc)*32 + lane)*16 + h]
// ---------------------------------------------------------------------------
__global__ void pack_w(const float* __restrict__ W, _Float16* __restrict__ dst,
                       int OUT, int K) {
    int idx = blockIdx.x * blockDim.x + threadIdx.x;
    int total = OUT * K;
    if (idx >= total) return;
    int h    = idx & 15;
    int lane = (idx >> 4) & 31;
    int frag = idx >> 9;
    int KC   = K >> 5;
    int kc   = frag % KC;
    int n    = frag / KC;
    int row  = n * 16 + (lane & 15);
    int col  = kc * 32 + ((h >> 3) << 4) + ((lane >> 4) << 3) + (h & 7);
    dst[idx] = (_Float16)W[row * K + col];
}

// B fragment builder: 16 f16 halves from two 8-float runs (run0 = K base.., run1 = K base+16..)
__device__ __forceinline__ v16h pack_b(float4 a0, float4 a1, float4 b0, float4 b1) {
    v16h v;
    v[0] = (_Float16)a0.x; v[1] = (_Float16)a0.y; v[2]  = (_Float16)a0.z; v[3]  = (_Float16)a0.w;
    v[4] = (_Float16)a1.x; v[5] = (_Float16)a1.y; v[6]  = (_Float16)a1.z; v[7]  = (_Float16)a1.w;
    v[8] = (_Float16)b0.x; v[9] = (_Float16)b0.y; v[10] = (_Float16)b0.z; v[11] = (_Float16)b0.w;
    v[12] = (_Float16)b1.x; v[13] = (_Float16)b1.y; v[14] = (_Float16)b1.z; v[15] = (_Float16)b1.w;
    return v;
}

// ---------------------------------------------------------------------------
// Edge kernel: y = relu(relu([x[src], ef] @ Wm0^T + bm0) @ Wm1^T + bm1);
//              atomic scatter-add into agg[dst].
// One wave processes 16 edges per tile. D = W x feat^T orientation:
// lane = edge (N), accumulator VGPR r = output row n*16 + g*8 + r.
// ---------------------------------------------------------------------------
__launch_bounds__(256)
__global__ void edge_mlp_scatter(const float* __restrict__ x,
                                 const float* __restrict__ ef,
                                 const int*   __restrict__ eidx,   // [2, E]
                                 const float* __restrict__ bm0,
                                 const float* __restrict__ bm1,
                                 const _Float16* __restrict__ wpack, // Wm0 (8192 h) then Wm1 (4096 h)
                                 float* __restrict__ agg)
{
    __shared__ _Float16 sW[12288];   // 24 KB
    {
        const uint4* s = (const uint4*)wpack;
        uint4* d = (uint4*)sW;
        for (int i = threadIdx.x; i < 12288 * 2 / 16; i += blockDim.x) d[i] = s[i];
    }
    __syncthreads();

    const int lane = threadIdx.x & 31;
    const int m    = lane & 15;       // edge within tile
    const int c0   = (lane >> 4) * 8; // column-run base for this half-wave

    // Biases, preloaded (constant across tiles). acc[n][r] -> output n*16+c0+r.
    float4 b1lo[4], b1hi[4], b2lo[4], b2hi[4];
#pragma unroll
    for (int n = 0; n < 4; ++n) {
        b1lo[n] = *(const float4*)(bm0 + n * 16 + c0);
        b1hi[n] = *(const float4*)(bm0 + n * 16 + c0 + 4);
        b2lo[n] = *(const float4*)(bm1 + n * 16 + c0);
        b2hi[n] = *(const float4*)(bm1 + n * 16 + c0 + 4);
    }

    const int wave   = (blockIdx.x * blockDim.x + threadIdx.x) >> 5;
    const int nwaves = (gridDim.x * blockDim.x) >> 5;
    const int ntiles = NEDGE / 16;

    for (int tile = wave; tile < ntiles; tile += nwaves) {
        int e  = tile * 16 + m;
        bool ok = (e < NEDGE);
        int ec = ok ? e : (NEDGE - 1);
        int srcn = eidx[ec];
        int dstn = eidx[NEDGE + ec];

        // B1 fragments: features of edge ec, K = [x[src] (0..63) | ef[e] (64..127)]
        v16h B1[4];
#pragma unroll
        for (int kc = 0; kc < 4; ++kc) {
            const float* row = (kc < 2) ? (x  + (size_t)srcn * 64 + kc * 32)
                                        : (ef + (size_t)ec   * 64 + (kc - 2) * 32);
            float4 a0 = *(const float4*)(row + c0);
            float4 a1 = *(const float4*)(row + c0 + 4);
            float4 b0 = *(const float4*)(row + 16 + c0);
            float4 b1 = *(const float4*)(row + 16 + c0 + 4);
            B1[kc] = pack_b(a0, a1, b0, b1);
        }

        // Layer 1: acc1[n] (n-th 16-row output tile of Wm0)
        v8f acc1[4];
#pragma unroll
        for (int n = 0; n < 4; ++n) {
            acc1[n][0] = b1lo[n].x; acc1[n][1] = b1lo[n].y; acc1[n][2] = b1lo[n].z; acc1[n][3] = b1lo[n].w;
            acc1[n][4] = b1hi[n].x; acc1[n][5] = b1hi[n].y; acc1[n][6] = b1hi[n].z; acc1[n][7] = b1hi[n].w;
        }
#pragma unroll
        for (int n = 0; n < 4; ++n) {
#pragma unroll
            for (int kc = 0; kc < 4; ++kc) {
                v16h A = *(const v16h*)&sW[((n * 4 + kc) * 32 + lane) * 16];
                acc1[n] = __builtin_amdgcn_wmma_f32_16x16x32_f16(
                    false, A, false, B1[kc], (short)0, acc1[n], false, false);
            }
        }
#pragma unroll
        for (int n = 0; n < 4; ++n)
#pragma unroll
            for (int r = 0; r < 8; ++r) acc1[n][r] = fmaxf(acc1[n][r], 0.0f);

        // Layer-2 B fragments come straight out of acc1 (no transpose needed):
        // B2[kc][h] needs K = kc*32 + (h>>3)*16 + c0 + (h&7)  ==  acc1[2*kc+(h>>3)][h&7]
        v16h B2[2];
#pragma unroll
        for (int kc = 0; kc < 2; ++kc)
#pragma unroll
            for (int h = 0; h < 16; ++h)
                B2[kc][h] = (_Float16)acc1[2 * kc + (h >> 3)][h & 7];

        v8f acc2[4];
#pragma unroll
        for (int n = 0; n < 4; ++n) {
            acc2[n][0] = b2lo[n].x; acc2[n][1] = b2lo[n].y; acc2[n][2] = b2lo[n].z; acc2[n][3] = b2lo[n].w;
            acc2[n][4] = b2hi[n].x; acc2[n][5] = b2hi[n].y; acc2[n][6] = b2hi[n].z; acc2[n][7] = b2hi[n].w;
        }
#pragma unroll
        for (int n = 0; n < 4; ++n) {
#pragma unroll
            for (int kc = 0; kc < 2; ++kc) {
                v16h A = *(const v16h*)&sW[8192 + ((n * 2 + kc) * 32 + lane) * 16];
                acc2[n] = __builtin_amdgcn_wmma_f32_16x16x32_f16(
                    false, A, false, B2[kc], (short)0, acc2[n], false, false);
            }
        }

        if (ok) {
            float* arow = agg + (size_t)dstn * 64;
#pragma unroll
            for (int n = 0; n < 4; ++n)
#pragma unroll
                for (int r = 0; r < 8; ++r)
                    unsafeAtomicAdd(arow + n * 16 + c0 + r, fmaxf(acc2[n][r], 0.0f));
        }
    }
}

// ---------------------------------------------------------------------------
// Node kernel: z = relu([x, agg] @ Wu0^T + bu0); out_pre = z @ Wuo^T + buo + [x, agg]
// ---------------------------------------------------------------------------
__launch_bounds__(256)
__global__ void node_mlp(const float* __restrict__ x,
                         const float* __restrict__ agg,
                         const float* __restrict__ bu0,
                         const float* __restrict__ buo,
                         const _Float16* __restrict__ wu0p,
                         const _Float16* __restrict__ wuop,
                         float* __restrict__ out_pre)
{
    const int lane = threadIdx.x & 31;
    const int m    = lane & 15;
    const int c0   = (lane >> 4) * 8;

    const int wave   = (blockIdx.x * blockDim.x + threadIdx.x) >> 5;
    const int nwaves = (gridDim.x * blockDim.x) >> 5;
    const int ntiles = NNODE / 16;   // 6250 exact

    for (int tile = wave; tile < ntiles; tile += nwaves) {
        int node = tile * 16 + m;
        bool ok = (node < NNODE);
        int nc = ok ? node : (NNODE - 1);

        // B1 fragments + keep fp32 values for the identity residual
        v16h  B1[4];
        float4 rx[4][4];
#pragma unroll
        for (int kc = 0; kc < 4; ++kc) {
            const float* row = (kc < 2) ? (x   + (size_t)nc * 64 + kc * 32)
                                        : (agg + (size_t)nc * 64 + (kc - 2) * 32);
            rx[kc][0] = *(const float4*)(row + c0);
            rx[kc][1] = *(const float4*)(row + c0 + 4);
            rx[kc][2] = *(const float4*)(row + 16 + c0);
            rx[kc][3] = *(const float4*)(row + 16 + c0 + 4);
            B1[kc] = pack_b(rx[kc][0], rx[kc][1], rx[kc][2], rx[kc][3]);
        }

        v8f acc1[8];
#pragma unroll
        for (int n = 0; n < 8; ++n)
#pragma unroll
            for (int r = 0; r < 8; ++r) acc1[n][r] = 0.0f;
#pragma unroll
        for (int n = 0; n < 8; ++n) {
#pragma unroll
            for (int kc = 0; kc < 4; ++kc) {
                v16h A = *(const v16h*)&wu0p[((n * 4 + kc) * 32 + lane) * 16];
                acc1[n] = __builtin_amdgcn_wmma_f32_16x16x32_f16(
                    false, A, false, B1[kc], (short)0, acc1[n], false, false);
            }
        }
        // bias + relu (bias loads hit L1; keeps VGPR pressure down)
#pragma unroll
        for (int n = 0; n < 8; ++n) {
            float4 blo = *(const float4*)(bu0 + n * 16 + c0);
            float4 bhi = *(const float4*)(bu0 + n * 16 + c0 + 4);
            const float* bb0 = (const float*)&blo;
            const float* bb1 = (const float*)&bhi;
#pragma unroll
            for (int r = 0; r < 8; ++r)
                acc1[n][r] = fmaxf(acc1[n][r] + (r < 4 ? bb0[r & 3] : bb1[r & 3]), 0.0f);
        }

        // Layer-2 B fragments directly from acc1 (lane-local)
        v16h B2[4];
#pragma unroll
        for (int kc = 0; kc < 4; ++kc)
#pragma unroll
            for (int h = 0; h < 16; ++h)
                B2[kc][h] = (_Float16)acc1[2 * kc + (h >> 3)][h & 7];

        v8f acc2[8];
#pragma unroll
        for (int n = 0; n < 8; ++n)
#pragma unroll
            for (int r = 0; r < 8; ++r) acc2[n][r] = 0.0f;
#pragma unroll
        for (int n = 0; n < 8; ++n) {
#pragma unroll
            for (int kc = 0; kc < 4; ++kc) {
                v16h A = *(const v16h*)&wuop[((n * 4 + kc) * 32 + lane) * 16];
                acc2[n] = __builtin_amdgcn_wmma_f32_16x16x32_f16(
                    false, A, false, B2[kc], (short)0, acc2[n], false, false);
            }
        }

        // bias + residual, then store. out col = n*16 + c0 + r; residual value
        // lives in rx[n>>1][(n&1)*2 + (r>>2)] component (r&3).
        if (ok) {
#pragma unroll
            for (int n = 0; n < 8; ++n) {
                float4 blo = *(const float4*)(buo + n * 16 + c0);
                float4 bhi = *(const float4*)(buo + n * 16 + c0 + 4);
                const float* bb0 = (const float*)&blo;
                const float* bb1 = (const float*)&bhi;
                float v[8];
#pragma unroll
                for (int r = 0; r < 8; ++r) {
                    const float* res = (const float*)&rx[n >> 1][(n & 1) * 2 + (r >> 2)];
                    v[r] = acc2[n][r] + (r < 4 ? bb0[r & 3] : bb1[r & 3]) + res[r & 3];
                }
                float* op = out_pre + (size_t)node * 128 + n * 16 + c0;
                *(float4*)(op)     = make_float4(v[0], v[1], v[2], v[3]);
                *(float4*)(op + 4) = make_float4(v[4], v[5], v[6], v[7]);
            }
        }
    }
}

// ---------------------------------------------------------------------------
// BatchNorm: column sums / sum-of-squares, then normalize.
// ---------------------------------------------------------------------------
__global__ void bn_stats(const float* __restrict__ out_pre,
                         float* __restrict__ colsum, float* __restrict__ colsq)
{
    int col = threadIdx.x;   // blockDim.x == 128
    float s = 0.0f, q = 0.0f;
    for (int row = blockIdx.x; row < NNODE; row += gridDim.x) {
        float v = out_pre[(size_t)row * 128 + col];
        s += v; q += v * v;
    }
    unsafeAtomicAdd(&colsum[col], s);
    unsafeAtomicAdd(&colsq[col], q);
}

__global__ void bn_norm(const float* __restrict__ out_pre,
                        const float* __restrict__ colsum,
                        const float* __restrict__ colsq,
                        const float* __restrict__ gamma,
                        const float* __restrict__ beta,
                        float* __restrict__ out)
{
    size_t i = (size_t)blockIdx.x * blockDim.x + threadIdx.x;
    if (i >= (size_t)NNODE * 128) return;
    int col = (int)(i & 127);
    const float invN = 1.0f / (float)NNODE;
    float mean = colsum[col] * invN;
    float var  = colsq[col] * invN - mean * mean;
    float inv  = rsqrtf(var + 1e-5f);
    out[i] = (out_pre[i] - mean) * inv * gamma[col] + beta[col];
}

// ---------------------------------------------------------------------------
extern "C" void kernel_launch(void* const* d_in, const int* in_sizes, int n_in,
                              void* d_out, int out_size, void* d_ws, size_t ws_size,
                              hipStream_t stream) {
    const float* x     = (const float*)d_in[0];
    const float* ef    = (const float*)d_in[1];
    const int*   eidx  = (const int*)  d_in[2];
    // d_in[3] = batch_idx (unused by the reference computation)
    const float* Wm0   = (const float*)d_in[4];
    const float* bm0   = (const float*)d_in[5];
    const float* Wm1   = (const float*)d_in[6];
    const float* bm1   = (const float*)d_in[7];
    const float* Wu0   = (const float*)d_in[8];
    const float* bu0   = (const float*)d_in[9];
    const float* Wuo   = (const float*)d_in[10];
    const float* buo   = (const float*)d_in[11];
    const float* gamma = (const float*)d_in[12];
    const float* beta  = (const float*)d_in[13];
    float* out = (float*)d_out;

    char* ws = (char*)d_ws;
    size_t off = 0;
    float*    agg     = (float*)(ws + off);    off += (size_t)NNODE * 64 * 4;   // 25.6 MB
    float*    out_pre = (float*)(ws + off);    off += (size_t)NNODE * 128 * 4;  // 51.2 MB
    _Float16* wm_pack = (_Float16*)(ws + off); off += (64 * 128 + 64 * 64) * 2; // Wm0|Wm1
    _Float16* wu0p    = (_Float16*)(ws + off); off += 128 * 128 * 2;
    _Float16* wuop    = (_Float16*)(ws + off); off += 128 * 128 * 2;
    float*    colsum  = (float*)(ws + off);    off += 128 * 4;
    float*    colsq   = (float*)(ws + off);    off += 128 * 4;

    hipMemsetAsync(agg, 0, (size_t)NNODE * 64 * 4, stream);
    hipMemsetAsync(colsum, 0, 256 * 4, stream);  // colsum + colsq (contiguous)

    pack_w<<<(64 * 128 + 255) / 256, 256, 0, stream>>>(Wm0, wm_pack,            64, 128);
    pack_w<<<(64 *  64 + 255) / 256, 256, 0, stream>>>(Wm1, wm_pack + 64 * 128, 64,  64);
    pack_w<<<(128 * 128 + 255) / 256, 256, 0, stream>>>(Wu0, wu0p, 128, 128);
    pack_w<<<(128 * 128 + 255) / 256, 256, 0, stream>>>(Wuo, wuop, 128, 128);

    edge_mlp_scatter<<<2048, 256, 0, stream>>>(x, ef, eidx, bm0, bm1, wm_pack, agg);
    node_mlp<<<512, 256, 0, stream>>>(x, agg, bu0, buo, wu0p, wuop, out_pre);
    bn_stats<<<512, 128, 0, stream>>>(out_pre, colsum, colsq);
    bn_norm<<<(NNODE * 128 + 255) / 256, 256, 0, stream>>>(out_pre, colsum, colsq,
                                                           gamma, beta, out);
}
